// GraphAttentionLayer_5437428597334
// MI455X (gfx1250) — matrix operand order repaired and checked by reading, not compiled, not run on hardware
//
#include <hip/hip_runtime.h>
#include <hip/hip_bf16.h>

typedef __attribute__((ext_vector_type(16))) _Float16 v16h;
typedef __attribute__((ext_vector_type(8)))  float    v8f;

#define BB    128
#define LL    512
#define FINN  128
#define FOUTN 128
#define MTOT  (BB * LL)     // 65536
#define ALPHA 0.2f

__device__ __forceinline__ void wait_async0() {
#if __has_builtin(__builtin_amdgcn_s_wait_asynccnt)
  __builtin_amdgcn_s_wait_asynccnt(0);
#else
  asm volatile("s_wait_asynccnt 0x0" ::: "memory");
#endif
}

// ---------------------------------------------------------------------------
// Kernel 1: Wh = h @ W (M=65536, N=128, K=128), f16 WMMA, f32 accumulate.
// Block: 256 threads = 8 waves; tile 128(M) x 16(N); wave owns 16x16.
// W n-slice staged in LDS TRANSPOSED [n][k] -> B fragment is ONE contiguous
// 32B v16h load. Output written transposed: Wh16T[b][n][l], one 16B store
// per lane (8 contiguous rows, fixed col).
// ---------------------------------------------------------------------------
__global__ __launch_bounds__(256) void gat_wh_gemm(
    const float* __restrict__ h, const float* __restrict__ W,
    _Float16* __restrict__ Wh16T)
{
  const int tid  = threadIdx.x;
  const int lane = tid & 31;
  const int wave = tid >> 5;       // 0..7
  const int hlf  = lane >> 4;      // 0/1
  const int l16  = lane & 15;

  const int n0 = blockIdx.y * 16;
  const int m0 = blockIdx.x * 128 + wave * 16;

  __shared__ __align__(32) _Float16 WtileT[16 * FINN];   // [n][k], 4 KB

  for (int idx = tid; idx < 16 * FINN; idx += 256) {
    const int k = idx >> 4, n = idx & 15;
    WtileT[n * FINN + k] = (_Float16)W[k * FOUTN + n0 + n];
  }
  __syncthreads();

  const int m = m0 + l16;                       // A row for this lane
  const float* __restrict__ hrow = h + (size_t)m * FINN;

  v8f acc = {};
  #pragma unroll
  for (int k0 = 0; k0 < FINN; k0 += 32) {
    // A fragment (16x32 f16): e=0..15 -> K=k0+16*hlf+e
    v16h a;
    const float4 c0 = *(const float4*)(hrow + k0 + 16 * hlf);
    const float4 c1 = *(const float4*)(hrow + k0 + 16 * hlf + 4);
    const float4 c2 = *(const float4*)(hrow + k0 + 16 * hlf + 8);
    const float4 c3 = *(const float4*)(hrow + k0 + 16 * hlf + 12);
    a[0]=(_Float16)c0.x; a[1]=(_Float16)c0.y; a[2]=(_Float16)c0.z; a[3]=(_Float16)c0.w;
    a[4]=(_Float16)c1.x; a[5]=(_Float16)c1.y; a[6]=(_Float16)c1.z; a[7]=(_Float16)c1.w;
    a[8]=(_Float16)c2.x; a[9]=(_Float16)c2.y; a[10]=(_Float16)c2.z; a[11]=(_Float16)c2.w;
    a[12]=(_Float16)c3.x; a[13]=(_Float16)c3.y; a[14]=(_Float16)c3.z; a[15]=(_Float16)c3.w;

    // B fragment: lane col n=l16, K = k0+16*hlf+e -> contiguous 16 halves
    const v16h bf = *(const v16h*)(WtileT + l16 * FINN + k0 + 16 * hlf);

    acc = __builtin_amdgcn_wmma_f32_16x16x32_f16(
        false, a, false, bf, (short)0, acc, false, false);
  }

  // D: element r -> row m0 + 8*hlf + r (contiguous), col n0 + l16 (fixed)
  {
    const int mi0 = m0 + 8 * hlf;
    const int b   = mi0 >> 9;
    const int l0  = mi0 & (LL - 1);
    const int n   = n0 + l16;
    _Float16 tmp[8];
    #pragma unroll
    for (int r = 0; r < 8; ++r) tmp[r] = (_Float16)acc[r];
    *(int4*)(Wh16T + (size_t)b * FOUTN * LL + (size_t)n * LL + l0) =
        *(const int4*)tmp;
  }
}

// ---------------------------------------------------------------------------
// Kernel 1b: ei/ej from transposed Wh16T (coalesced over l, loop over n).
// ---------------------------------------------------------------------------
__global__ __launch_bounds__(256) void gat_eiej(
    const _Float16* __restrict__ Wh16T, const float* __restrict__ a,
    float* __restrict__ ei, float* __restrict__ ej)
{
  const int b   = blockIdx.x;
  const int tid = threadIdx.x;
  const _Float16* __restrict__ base = Wh16T + (size_t)b * FOUTN * LL;
  float si0 = 0.f, sj0 = 0.f, si1 = 0.f, sj1 = 0.f;
  #pragma unroll 4
  for (int n = 0; n < FOUTN; ++n) {
    const float ai = a[n], aj = a[FOUTN + n];
    const float w0 = (float)base[(size_t)n * LL + tid];
    const float w1 = (float)base[(size_t)n * LL + tid + 256];
    si0 += w0 * ai; sj0 += w0 * aj;
    si1 += w1 * ai; sj1 += w1 * aj;
  }
  ei[b * LL + tid]       = si0;
  ej[b * LL + tid]       = sj0;
  ei[b * LL + tid + 256] = si1;
  ej[b * LL + tid + 256] = sj1;
}

// ---------------------------------------------------------------------------
// Issue async global->LDS copies of one 32(k) x 128(n) Wh tile, stored
// TRANSPOSED as whS[n][k] (8 KB). Each thread issues two 16B transfers.
// Tracked by ASYNCcnt; caller waits with s_wait_asynccnt + barrier.
// ---------------------------------------------------------------------------
__device__ __forceinline__ void stage_wh_async(
    const _Float16* __restrict__ whB, unsigned whS_base,
    int buf, int k0, int tid)
{
  #pragma unroll
  for (int i = 0; i < 2; ++i) {
    const int c = tid + i * 256;          // chunk 0..511
    const int n = c >> 2, q = c & 3;      // 4 x 16B chunks per n-row
    const unsigned lds  = whS_base + (unsigned)(buf * 8192 + c * 16);
    const unsigned voff = (unsigned)(n * (LL * 2) + k0 * 2 + q * 16);
    asm volatile("global_load_async_to_lds_b128 %0, %1, %2"
                 :: "v"(lds), "v"(voff), "s"(whB)
                 : "memory");
  }
}

// ---------------------------------------------------------------------------
// Kernel 2: logits + softmax (register-resident, 2 rows/wave) + WMMA
// (att 16x512 @ Wh 512x128) with async double-buffered Wh tiles + ELU.
// ---------------------------------------------------------------------------
__global__ __launch_bounds__(256) void gat_attn(
    const _Float16* __restrict__ Wh16T,
    const float* __restrict__ ei, const float* __restrict__ ej,
    const float* __restrict__ edge_bias,
    const unsigned char* __restrict__ adj_mask,
    float* __restrict__ out)
{
  const int b    = blockIdx.y;
  const int i0   = blockIdx.x * 16;
  const int tid  = threadIdx.x;
  const int lane = tid & 31;
  const int wave = tid >> 5;
  const int hlf  = lane >> 4;
  const int l16  = lane & 15;

  __shared__ __align__(16) _Float16 attS[16 * LL];        // 16 KB
  __shared__ __align__(32) _Float16 whS[2][FOUTN * 32];   // 16 KB, [n][k] x2
  __shared__ float rowsum[16];

  // --- 1. fused logits + softmax, 2 rows per wave, logits in registers ---
  const float* __restrict__ ebB = edge_bias + ((size_t)b * LL + i0) * LL;
  #pragma unroll
  for (int rr = 0; rr < 2; ++rr) {
    const int ir = wave * 2 + rr;
    const float eival = ei[b * LL + i0 + ir];
    float ev[16];
    float mx = -3.0e38f;
    #pragma unroll
    for (int t = 0; t < 16; ++t) {
      const int j = lane + t * 32;
      float v = eival + ej[b * LL + j];
      v = (v > 0.f) ? v : ALPHA * v;
      v += ebB[(size_t)ir * LL + j];
      v = (adj_mask[(size_t)(i0 + ir) * LL + j] != 0) ? v : -3.0e38f;
      ev[t] = v;
      mx = fmaxf(mx, v);
    }
    #pragma unroll
    for (int off = 16; off; off >>= 1) mx = fmaxf(mx, __shfl_xor(mx, off, 32));
    float s = 0.f;
    #pragma unroll
    for (int t = 0; t < 16; ++t) {
      const float p = __expf(ev[t] - mx);               // masked -> 0
      attS[ir * LL + lane + t * 32] = (_Float16)p;
      s += p;
    }
    #pragma unroll
    for (int off = 16; off; off >>= 1) s += __shfl_xor(s, off, 32);
    if (lane == 0) rowsum[ir] = s;
  }

  // --- 2. GEMM att(16x512) x Wh16[b](512x128), async double-buffered ---
  const _Float16* __restrict__ whB = Wh16T + (size_t)b * FOUTN * LL;
  const unsigned whS_base = (unsigned)(uintptr_t)(&whS[0][0]);

  stage_wh_async(whB, whS_base, 0, 0, tid);   // prologue: tile 0 -> buf 0
  wait_async0();
  __syncthreads();          // also publishes attS / rowsum

  v8f acc = {};
  for (int k0 = 0; k0 < LL; k0 += 32) {
    const int cur = (k0 >> 5) & 1;
    if (k0 + 32 < LL)                         // prefetch next tile
      stage_wh_async(whB, whS_base, cur ^ 1, k0 + 32, tid);

    // A fragment: row l16; K = 16*hlf+e -> contiguous 16 halves in attS
    v16h a;
    {
      const _Float16* ap = attS + l16 * LL + k0 + 16 * hlf;
      ((int4*)&a)[0] = *(const int4*)ap;          // e=0..7
      ((int4*)&a)[1] = *(const int4*)(ap + 8);    // e=8..15
    }
    // B fragment: col n = 16*wave + l16; K = 16*hlf+e -> contiguous
    const v16h bf = *(const v16h*)(&whS[cur][(wave * 16 + l16) * 32 + 16 * hlf]);

    acc = __builtin_amdgcn_wmma_f32_16x16x32_f16(
        false, a, false, bf, (short)0, acc, false, false);

    wait_async0();          // own async writes to buf cur^1 complete
    __syncthreads();        // publish buf cur^1; buf cur safe to overwrite next
  }

  // --- 3. normalize, ELU, store ---
  #pragma unroll
  for (int r = 0; r < 8; ++r) {
    const int ir = r + 8 * hlf;
    float v = acc[r] / rowsum[ir];
    v = (v > 0.f) ? v : (__expf(v) - 1.f);
    out[((size_t)b * LL + i0 + ir) * FOUTN + wave * 16 + l16] = v;
  }
}

// ---------------------------------------------------------------------------
extern "C" void kernel_launch(void* const* d_in, const int* in_sizes, int n_in,
                              void* d_out, int out_size, void* d_ws, size_t ws_size,
                              hipStream_t stream)
{
  const float*         h    = (const float*)d_in[0];         // [B,L,FIN]
  const float*         W    = (const float*)d_in[1];         // [FIN,FOUT]
  const float*         a    = (const float*)d_in[2];         // [2*FOUT]
  const float*         eb   = (const float*)d_in[3];         // [B,L,L]
  const unsigned char* mask = (const unsigned char*)d_in[4]; // [L,L] bool
  float*               out  = (float*)d_out;                 // [B,L,FOUT]

  char* ws = (char*)d_ws;
  const size_t whf16_bytes = (size_t)MTOT * FOUTN * sizeof(_Float16); // 16 MB
  _Float16* Wh16T = (_Float16*)ws;                   // [B][FOUT][L]
  float*    ei    = (float*)(ws + whf16_bytes);
  float*    ej    = ei + MTOT;

  dim3 g1(MTOT / 128, FOUTN / 16);   // 512 x 8
  gat_wh_gemm<<<g1, 256, 0, stream>>>(h, W, Wh16T);

  gat_eiej<<<BB, 256, 0, stream>>>(Wh16T, a, ei, ej);

  dim3 g2(LL / 16, BB);              // 32 x 128
  gat_attn<<<g2, 256, 0, stream>>>(Wh16T, ei, ej, eb, mask, out);
}
